// STA_Block_28492813041796
// MI455X (gfx1250) — matrix operand order repaired and checked by reading, not compiled, not run on hardware
//
#include <hip/hip_runtime.h>
#include <hip/hip_bf16.h>

typedef _Float16 half_t;
typedef _Float16 v8h  __attribute__((ext_vector_type(8)));
typedef _Float16 v16h __attribute__((ext_vector_type(16)));
typedef float    v8f  __attribute__((ext_vector_type(8)));
typedef int      v2i  __attribute__((vector_size(8)));

#define NN 16
#define CC 128
#define TT 120
#define VV 25
#define HH 8
#define DD 32
#define POS 3000          // T*V
#define PP  3072          // padded position count (multiple of 128)

// ---------------------------------------------------------------------------
// CDNA5 async global->LDS copy (ASYNCcnt path), guarded so absence of the
// builtin falls back to plain VGPR staging. The builtin types its pointers
// by the transfer width: b64 -> pointer-to-v2i32 (global / LDS).
// ---------------------------------------------------------------------------
#if defined(__has_builtin)
#if __has_builtin(__builtin_amdgcn_global_load_async_to_lds_b64) && \
    __has_builtin(__builtin_amdgcn_s_wait_asynccnt)
#define HAVE_ASYNC_LDS 1
#endif
#endif

#ifdef HAVE_ASYNC_LDS
__device__ __forceinline__ void async_cp_b64(void* lds, const void* gptr) {
  __builtin_amdgcn_global_load_async_to_lds_b64(
      (__attribute__((address_space(1))) v2i*)(gptr),
      (__attribute__((address_space(3))) v2i*)(lds), 0, 0);
}
__device__ __forceinline__ void async_wait() {
  __builtin_amdgcn_s_wait_asynccnt(0);
}
#endif

// ---------------------------------------------------------------------------
// WMMA fragment helpers (CDNA5 wave32, 16x16x32 f16 -> f32)
// A-matrix 16x32: lane<16 -> M=lane, halves 0..7 = K 0..7, halves 8..15 = K16..23
//                 lane>=16 -> M=lane-16, K base +8 (per ISA 7.12.2)
// B-matrix 32x16: lane%16 = N col, halves 0..7 = K 0..7, halves 8..15 = K16..23 (lane>=16: K base +8)
// ---------------------------------------------------------------------------
__device__ __forceinline__ v16h frag_A(const half_t* base, int ld) {
  const int lane = threadIdx.x & 31;
  const int row  = lane & 15;
  const int hi   = lane >> 4;
  const half_t* p = base + row * ld + hi * 8;
  union { v16h v; v8h h[2]; } u;
  u.h[0] = *(const v8h*)(p);
  u.h[1] = *(const v8h*)(p + 16);
  return u.v;
}
__device__ __forceinline__ v16h frag_B(const half_t* baseT, int ld) {
  // baseT laid out [N][K]
  const int lane = threadIdx.x & 31;
  const int col  = lane & 15;
  const int hi   = lane >> 4;
  const half_t* p = baseT + col * ld + hi * 8;
  union { v16h v; v8h h[2]; } u;
  u.h[0] = *(const v8h*)(p);
  u.h[1] = *(const v8h*)(p + 16);
  return u.v;
}
#define WMMA_F16(a, b, c) \
  __builtin_amdgcn_wmma_f32_16x16x32_f16(false, (a), false, (b), (short)0, (c), false, false)

// ---------------------------------------------------------------------------
// Kernel 0: pack fp32 x (N,C,3000) -> f16 padded (N*C, PP), zero-fill padding
// ---------------------------------------------------------------------------
__global__ void k_pack_x(const float* __restrict__ x, half_t* __restrict__ xh) {
  const int row = blockIdx.x;                 // n*128 + c, 0..2047
  const float* src = x + (size_t)row * POS;
  half_t* dst = xh + (size_t)row * PP;
  for (int col = threadIdx.x; col < PP; col += blockDim.x)
    dst[col] = (col < POS) ? (half_t)src[col] : (half_t)0.f;
}

// ---------------------------------------------------------------------------
// Generic conv GEMM: Out(M x PP) = W(M x K) * Bh(K x PP) + bias, per batch n.
// Optional fused epilogue: BN -> +residual -> leaky ReLU.
// Block: 256 thr (8 waves, 2x4), tile 64(M) x 128(N), K-step 32.
// Each wave: 32x32 tile -> 4 WMMA per K-step.
// ---------------------------------------------------------------------------
__global__ __launch_bounds__(256)
void k_gemm_conv(const float* __restrict__ W, const half_t* __restrict__ Bh,
                 const float* __restrict__ bias, int M, int K,
                 const float* __restrict__ bn_g, const float* __restrict__ bn_b,
                 const float* __restrict__ bn_m, const float* __restrict__ bn_v,
                 const float* __restrict__ res, size_t res_n_stride, int res_ld,
                 half_t* __restrict__ outH,
                 float* __restrict__ outF, size_t outF_n_stride, int outF_ld,
                 int outF_pad) {
  __shared__ __attribute__((aligned(16))) half_t As[64][40];
  __shared__ __attribute__((aligned(16))) half_t Bt[128][40];  // [N][K]
  const int n  = blockIdx.z;
  const int m0 = blockIdx.y * 64;
  const int n0 = blockIdx.x * 128;
  const int tid = threadIdx.x, lane = tid & 31, wid = tid >> 5;
  const int wm = wid >> 2, wn = wid & 3;       // 2 x 4 waves
  const half_t* Bn = Bh + (size_t)n * K * PP;

  v8f acc[2][2] = {};

  for (int k0 = 0; k0 < K; k0 += 32) {
    if (k0 + 32 < K)   // prefetch next B tile (global_prefetch_b8)
      __builtin_prefetch(Bn + (size_t)(k0 + 32) * PP + n0 + (tid & 127), 0, 0);
#pragma unroll
    for (int e = 0; e < 8; ++e) {            // stage A, f32 -> f16 (2048 elts)
      int idx = tid + 256 * e;
      int r = idx >> 5, kk = idx & 31;
      As[r][kk] = (half_t)W[(size_t)(m0 + r) * K + (k0 + kk)];
    }
#pragma unroll
    for (int e = 0; e < 16; ++e) {           // stage B transposed (4096 elts)
      int idx = tid + 256 * e;
      int col = idx & 127, kk = idx >> 7;
      Bt[col][kk] = Bn[(size_t)(k0 + kk) * PP + (n0 + col)];
    }
    __syncthreads();
    v16h a0 = frag_A(&As[wm * 32][0], 40);
    v16h a1 = frag_A(&As[wm * 32 + 16][0], 40);
    v16h b0 = frag_B(&Bt[wn * 32][0], 40);
    v16h b1 = frag_B(&Bt[wn * 32 + 16][0], 40);
    acc[0][0] = WMMA_F16(a0, b0, acc[0][0]);
    acc[0][1] = WMMA_F16(a0, b1, acc[0][1]);
    acc[1][0] = WMMA_F16(a1, b0, acc[1][0]);
    acc[1][1] = WMMA_F16(a1, b1, acc[1][1]);
    __syncthreads();
  }

  const int hi = lane >> 4;
#pragma unroll
  for (int mi = 0; mi < 2; ++mi)
#pragma unroll
    for (int ni = 0; ni < 2; ++ni) {
      const int col = n0 + wn * 32 + ni * 16 + (lane & 15);
#pragma unroll
      for (int r = 0; r < 8; ++r) {
        const int row = m0 + wm * 32 + mi * 16 + hi * 8 + r;
        float v = acc[mi][ni][r] + bias[row];
        if (bn_g) {
          float s = bn_g[row] * rsqrtf(bn_v[row] + 1e-5f);
          v = (v - bn_m[row]) * s + bn_b[row];
          float rr = (col < POS || res_ld == PP)
                       ? res[(size_t)n * res_n_stride + (size_t)row * res_ld + col] : 0.f;
          v += rr;
          v = (v >= 0.f) ? v : 0.1f * v;
        }
        if (outH) outH[(size_t)n * M * PP + (size_t)row * PP + col] = (half_t)v;
        if (outF) {
          if (outF_pad || col < POS)
            outF[(size_t)n * outF_n_stride + (size_t)row * outF_ld + col] = v;
        }
      }
    }
}

// ---------------------------------------------------------------------------
// Spatial attention: att[n,h,u,v] = tanh(sum_{d,t} q*k / (D*T))*alpha + att0
// M=N=25 (pad 32), K=3840 split over 4 waves; LDS reduction.
// Output stored TRANSPOSED as attT[v][u] (32x32 f16) so the consumer's
// [N][K] LDS tile is a contiguous copy (async-DMA friendly).
// ---------------------------------------------------------------------------
__global__ __launch_bounds__(128)
void k_att_s(const half_t* __restrict__ qk, const float* __restrict__ alphas,
             const float* __restrict__ att0, half_t* __restrict__ attOut) {
  __shared__ __attribute__((aligned(16))) half_t As[4][32][40];
  __shared__ __attribute__((aligned(16))) half_t Bs[4][32][40];
  __shared__ float Red[4][32][32];
  const int h = blockIdx.x, n = blockIdx.y;
  const int tid = threadIdx.x, lane = tid & 31, w = tid >> 5;
  const half_t* qn = qk + (size_t)n * 512 * PP + (size_t)(h * DD) * PP;
  const half_t* kn = qn + (size_t)256 * PP;
  const int u = lane;
  v8f acc[2][2] = {};
  for (int step = 0; step < 30; ++step) {
    const int kbase = w * 960 + step * 32;
#pragma unroll
    for (int j = 0; j < 32; ++j) {
      int kk = kbase + j;
      int d = kk / TT, t = kk % TT;
      size_t off = (size_t)d * PP + t * VV + u;
      As[w][u][j] = (u < VV) ? qn[off] : (half_t)0.f;
      Bs[w][u][j] = (u < VV) ? kn[off] : (half_t)0.f;
    }
    v16h a0 = frag_A(&As[w][0][0], 40);
    v16h a1 = frag_A(&As[w][16][0], 40);
    v16h b0 = frag_B(&Bs[w][0][0], 40);
    v16h b1 = frag_B(&Bs[w][16][0], 40);
    acc[0][0] = WMMA_F16(a0, b0, acc[0][0]);
    acc[0][1] = WMMA_F16(a0, b1, acc[0][1]);
    acc[1][0] = WMMA_F16(a1, b0, acc[1][0]);
    acc[1][1] = WMMA_F16(a1, b1, acc[1][1]);
  }
  const int hi = lane >> 4;
#pragma unroll
  for (int mi = 0; mi < 2; ++mi)
#pragma unroll
    for (int ni = 0; ni < 2; ++ni)
#pragma unroll
      for (int r = 0; r < 8; ++r)
        Red[w][mi * 16 + hi * 8 + r][ni * 16 + (lane & 15)] = acc[mi][ni][r];
  __syncthreads();
  half_t* ob = attOut + ((size_t)(n * HH + h)) * 1024;
  for (int idx = tid; idx < 1024; idx += 128) {
    int uu = idx & 31, vv = idx >> 5;          // store addr = vv*32+uu = idx
    float s = Red[0][uu][vv] + Red[1][uu][vv] + Red[2][uu][vv] + Red[3][uu][vv];
    float val = 0.f;
    if (uu < VV && vv < VV)
      val = tanhf(s * (1.0f / 3840.0f)) * alphas[h] + att0[h * 625 + uu * 25 + vv];
    ob[idx] = (half_t)val;                     // attT[v][u]
  }
}

// ---------------------------------------------------------------------------
// Temporal attention: att[n,h,t,q] = tanh(sum_{d,v} q*k / (D*V))*alpha + att0
// M=N=120 (pad 128), K = 32 d-steps x 32 (v padded). One block per (n,h).
// Output stored TRANSPOSED as attT[q][t] (128x128 f16).
// ---------------------------------------------------------------------------
__global__ __launch_bounds__(256)
void k_att_t(const half_t* __restrict__ qk, const float* __restrict__ alphat,
             const float* __restrict__ att0, half_t* __restrict__ attOut) {
  __shared__ __attribute__((aligned(16))) half_t As[128][40];
  __shared__ __attribute__((aligned(16))) half_t Bt[128][40];
  const int h = blockIdx.x, n = blockIdx.y;
  const int tid = threadIdx.x, lane = tid & 31, wid = tid >> 5;
  const int wm = wid >> 1, wn = wid & 1;
  const half_t* qn = qk + (size_t)n * 512 * PP + (size_t)(h * DD) * PP;
  const half_t* kn = qn + (size_t)256 * PP;
  v8f acc[2][4] = {};
  for (int d = 0; d < DD; ++d) {
    const half_t* qd = qn + (size_t)d * PP;
    const half_t* kd = kn + (size_t)d * PP;
#pragma unroll
    for (int e = 0; e < 16; ++e) {
      int idx = tid + 256 * e;               // 4096 elements
      int row = idx >> 5, j = idx & 31;
      bool ok = (j < VV) && (row < TT);
      As[row][j] = ok ? qd[row * VV + j] : (half_t)0.f;
      Bt[row][j] = ok ? kd[row * VV + j] : (half_t)0.f;
    }
    __syncthreads();
#pragma unroll
    for (int mi = 0; mi < 2; ++mi) {
      v16h a = frag_A(&As[wm * 32 + mi * 16][0], 40);
#pragma unroll
      for (int ni = 0; ni < 4; ++ni) {
        v16h b = frag_B(&Bt[wn * 64 + ni * 16][0], 40);
        acc[mi][ni] = WMMA_F16(a, b, acc[mi][ni]);
      }
    }
    __syncthreads();
  }
  const int hi = lane >> 4;
  half_t* out = attOut + ((size_t)(n * HH + h)) * 16384;
#pragma unroll
  for (int mi = 0; mi < 2; ++mi)
#pragma unroll
    for (int ni = 0; ni < 4; ++ni)
#pragma unroll
      for (int r = 0; r < 8; ++r) {
        int t = wm * 32 + mi * 16 + hi * 8 + r;
        int q = wn * 64 + ni * 16 + (lane & 15);
        float val = 0.f;
        if (t < TT && q < TT)
          val = tanhf(acc[mi][ni][r] * (1.0f / 800.0f)) * alphat[h] +
                att0[h * 14400 + t * 120 + q];
        out[q * 128 + t] = (half_t)val;        // attT[q][t]
      }
}

// ---------------------------------------------------------------------------
// Spatial mix: xs[n,h,c,t,v] = sum_u x[c,t,u] * att[u,v].
// One block per (t,h,n): 128(c) x 32(v) x 32(u) WMMA.
// B tile = attT[v][u], contiguous -> async global->LDS copy.
// ---------------------------------------------------------------------------
__global__ __launch_bounds__(256)
void k_mix_s(const half_t* __restrict__ xh, const half_t* __restrict__ attT,
             half_t* __restrict__ outH) {
  __shared__ __attribute__((aligned(16))) half_t As[128][40];
  __shared__ __attribute__((aligned(16))) half_t Bt[32][40];
  const int t = blockIdx.x, h = blockIdx.y, n = blockIdx.z;
  const int tid = threadIdx.x, lane = tid & 31, w = tid >> 5;
  const half_t* xn = xh + (size_t)n * CC * PP;
  const half_t* an = attT + ((size_t)(n * HH + h)) * 1024;   // [v][u] 32x32
#ifdef HAVE_ASYNC_LDS
  {                                            // 1024 halves = 256 b64 chunks
    int v = tid >> 3, u0 = (tid & 7) * 4;
    async_cp_b64(&Bt[v][u0], an + v * 32 + u0);
  }
#else
#pragma unroll
  for (int e = 0; e < 4; ++e) {
    int idx = tid + 256 * e;                   // 1024
    int u = idx & 31, v = idx >> 5;
    Bt[v][u] = an[v * 32 + u];
  }
#endif
#pragma unroll
  for (int e = 0; e < 16; ++e) {
    int idx = tid + 256 * e;                   // 4096
    int c = idx >> 5, u = idx & 31;
    As[c][u] = (u < VV) ? xn[(size_t)c * PP + t * VV + u] : (half_t)0.f;
  }
#ifdef HAVE_ASYNC_LDS
  async_wait();
#endif
  __syncthreads();
  v16h a  = frag_A(&As[w * 16][0], 40);
  v16h b0 = frag_B(&Bt[0][0], 40);
  v16h b1 = frag_B(&Bt[16][0], 40);
  v8f acc0 = {}, acc1 = {};
  acc0 = WMMA_F16(a, b0, acc0);
  acc1 = WMMA_F16(a, b1, acc1);
  const int hi = lane >> 4;
  half_t* on = outH + (size_t)n * 1024 * PP + (size_t)(h * CC) * PP;
#pragma unroll
  for (int sub = 0; sub < 2; ++sub) {
    v8f acc = sub ? acc1 : acc0;
    int v = sub * 16 + (lane & 15);
#pragma unroll
    for (int r = 0; r < 8; ++r) {
      int c = w * 16 + hi * 8 + r;
      if (v < VV) on[(size_t)c * PP + t * VV + v] = (half_t)acc[r];
    }
  }
}

// ---------------------------------------------------------------------------
// Temporal mix: xt[n,h,c,v,q] = sum_t y[c,t,v] * att[t,q].
// Rows (c*25+v), 3200 total; block tile 64 x 128, K=128 (t padded).
// B tile = attT[q][t], contiguous -> async global->LDS copy.
// ---------------------------------------------------------------------------
__global__ __launch_bounds__(256)
void k_mix_t(const half_t* __restrict__ yh, const half_t* __restrict__ attT,
             half_t* __restrict__ outH) {
  __shared__ __attribute__((aligned(16))) half_t As[64][40];
  __shared__ __attribute__((aligned(16))) half_t Bt[128][40];
  const int m0 = blockIdx.x * 64;              // 0..3136
  const int h = blockIdx.y, n = blockIdx.z;
  const int tid = threadIdx.x, lane = tid & 31, wid = tid >> 5;
  const int wm = wid >> 1, wn = wid & 1;
  const half_t* yn = yh + (size_t)n * CC * PP;
  const half_t* an = attT + ((size_t)(n * HH + h)) * 16384;  // [q][t] 128x128
  v8f acc[4] = {};
  for (int k0 = 0; k0 < 128; k0 += 32) {
#ifdef HAVE_ASYNC_LDS
#pragma unroll
    for (int e = 0; e < 4; ++e) {              // 4096 halves = 1024 b64 chunks
      int chunk = tid + 256 * e;
      int q = chunk >> 3, j0 = (chunk & 7) * 4;
      async_cp_b64(&Bt[q][j0], an + q * 128 + k0 + j0);
    }
#else
#pragma unroll
    for (int e = 0; e < 16; ++e) {             // 4096 elements
      int idx = tid + 256 * e;
      int j = idx & 31, q = idx >> 5;
      Bt[q][j] = an[q * 128 + k0 + j];
    }
#endif
#pragma unroll
    for (int e = 0; e < 8; ++e) {              // 2048 elements
      int idx = tid + 256 * e;
      int r = idx >> 5, j = idx & 31;
      int row = m0 + r;
      int c = row / VV, v = row % VV;
      int t = k0 + j;
      As[r][j] = (t < TT) ? yn[(size_t)c * PP + t * VV + v] : (half_t)0.f;
    }
#ifdef HAVE_ASYNC_LDS
    async_wait();
#endif
    __syncthreads();
    v16h a = frag_A(&As[wm * 16][0], 40);
#pragma unroll
    for (int ni = 0; ni < 4; ++ni) {
      v16h b = frag_B(&Bt[wn * 64 + ni * 16][0], 40);
      acc[ni] = WMMA_F16(a, b, acc[ni]);
    }
    __syncthreads();
  }
  const int hi = lane >> 4;
  half_t* on = outH + (size_t)n * 1024 * PP + (size_t)(h * CC) * PP;
#pragma unroll
  for (int ni = 0; ni < 4; ++ni)
#pragma unroll
    for (int r = 0; r < 8; ++r) {
      int row = m0 + wm * 16 + hi * 8 + r;
      int c = row / VV, v = row % VV;
      int q = wn * 64 + ni * 16 + (lane & 15);
      if (q < TT) on[(size_t)c * PP + v * TT + q] = (half_t)acc[ni][r];
    }
}

// ---------------------------------------------------------------------------
// Workspace layout
// ---------------------------------------------------------------------------
static const size_t SZ_XH   = (size_t)NN * CC * PP * 2;      // f16 x
static const size_t SZ_YH   = SZ_XH;                         // f16 y
static const size_t SZ_YF   = (size_t)NN * CC * PP * 4;      // f32 y (padded)
static const size_t SZ_QK   = (size_t)NN * 512 * PP * 2;     // f16 qkv out
static const size_t SZ_MID  = (size_t)NN * 1024 * PP * 2;    // f16 xs/xt
static const size_t SZ_S4   = SZ_XH;                         // f16 stage-4 out
static const size_t SZ_ATTS = (size_t)128 * 1024 * 2;        // f16 attT_s 32x32
static const size_t SZ_ATTT = (size_t)128 * 16384 * 2;       // f16 attT_t 128x128

extern "C" void kernel_launch(void* const* d_in, const int* in_sizes, int n_in,
                              void* d_out, int out_size, void* d_ws, size_t ws_size,
                              hipStream_t stream) {
  (void)in_sizes; (void)n_in; (void)out_size;
  const float* x      = (const float*)d_in[0];
  const float* Wqkvs  = (const float*)d_in[1];
  const float* bqkvs  = (const float*)d_in[2];
  const float* alphas = (const float*)d_in[3];
  const float* att0s  = (const float*)d_in[4];
  const float* Wouts  = (const float*)d_in[5];
  const float* bouts  = (const float*)d_in[6];
  const float* gs1 = (const float*)d_in[7],  *bs1 = (const float*)d_in[8];
  const float* ms1 = (const float*)d_in[9],  *vs1 = (const float*)d_in[10];
  const float* Wffs = (const float*)d_in[11], *bffs = (const float*)d_in[12];
  const float* gs2 = (const float*)d_in[13], *bs2 = (const float*)d_in[14];
  const float* ms2 = (const float*)d_in[15], *vs2 = (const float*)d_in[16];
  const float* Wqkvt  = (const float*)d_in[17];
  const float* bqkvt  = (const float*)d_in[18];
  const float* alphat = (const float*)d_in[19];
  const float* att0t  = (const float*)d_in[20];
  const float* Woutt  = (const float*)d_in[21];
  const float* boutt  = (const float*)d_in[22];
  const float* gt1 = (const float*)d_in[23], *bt1 = (const float*)d_in[24];
  const float* mt1 = (const float*)d_in[25], *vt1 = (const float*)d_in[26];
  const float* Wfft = (const float*)d_in[27], *bfft = (const float*)d_in[28];
  const float* gt2 = (const float*)d_in[29], *bt2 = (const float*)d_in[30];
  const float* mt2 = (const float*)d_in[31], *vt2 = (const float*)d_in[32];

  char* ws = (char*)d_ws;
  size_t off = 0;
  half_t* xh  = (half_t*)(ws + off); off += SZ_XH;
  half_t* yh  = (half_t*)(ws + off); off += SZ_YH;
  float*  yf  = (float*)(ws + off);  off += SZ_YF;
  half_t* qk  = (half_t*)(ws + off); off += SZ_QK;
  half_t* mid = (half_t*)(ws + off); off += SZ_MID;
  half_t* s4  = (half_t*)(ws + off); off += SZ_S4;
  half_t* atts = (half_t*)(ws + off); off += SZ_ATTS;
  half_t* attt = (half_t*)(ws + off); off += SZ_ATTT;
  const size_t ws_needed = off;

  (void)hipMemsetAsync(d_ws, 0, ws_needed <= ws_size ? ws_needed : ws_size, stream);
  k_pack_x<<<NN * CC, 256, 0, stream>>>(x, xh);

  // ---- spatial branch ----
  k_gemm_conv<<<dim3(PP / 128, 8, NN), 256, 0, stream>>>(
      Wqkvs, xh, bqkvs, 512, 128,
      nullptr, nullptr, nullptr, nullptr, nullptr, 0, 0,
      qk, nullptr, 0, 0, 0);
  k_att_s<<<dim3(HH, NN), 128, 0, stream>>>(qk, alphas, att0s, atts);
  k_mix_s<<<dim3(TT, HH, NN), 256, 0, stream>>>(xh, atts, mid);
  k_gemm_conv<<<dim3(PP / 128, 2, NN), 256, 0, stream>>>(
      Wouts, mid, bouts, 128, 1024,
      gs1, bs1, ms1, vs1, x, (size_t)CC * POS, POS,
      s4, nullptr, 0, 0, 0);
  k_gemm_conv<<<dim3(PP / 128, 2, NN), 256, 0, stream>>>(
      Wffs, s4, bffs, 128, 128,
      gs2, bs2, ms2, vs2, x, (size_t)CC * POS, POS,
      yh, yf, (size_t)CC * PP, PP, 1);

  // ---- temporal branch ----
  k_gemm_conv<<<dim3(PP / 128, 8, NN), 256, 0, stream>>>(
      Wqkvt, yh, bqkvt, 512, 128,
      nullptr, nullptr, nullptr, nullptr, nullptr, 0, 0,
      qk, nullptr, 0, 0, 0);
  k_att_t<<<dim3(HH, NN), 256, 0, stream>>>(qk, alphat, att0t, attt);
  k_mix_t<<<dim3(50, HH, NN), 256, 0, stream>>>(yh, attt, mid);
  k_gemm_conv<<<dim3(PP / 128, 2, NN), 256, 0, stream>>>(
      Woutt, mid, boutt, 128, 1024,
      gt1, bt1, mt1, vt1, yf, (size_t)CC * PP, PP,
      s4, nullptr, 0, 0, 0);
  k_gemm_conv<<<dim3(PP / 128, 2, NN), 256, 0, stream>>>(
      Wfft, s4, bfft, 128, 128,
      gt2, bt2, mt2, vt2, yf, (size_t)CC * PP, PP,
      nullptr, (float*)d_out, (size_t)CC * POS, POS, 0);
}
